// MhAttnBlock_7627861918129
// MI455X (gfx1250) — matrix operand order repaired and checked by reading, compiled, not run on hardware
//
#include <hip/hip_runtime.h>
#include <hip/hip_bf16.h>

typedef __attribute__((ext_vector_type(16))) _Float16 v16h;
typedef __attribute__((ext_vector_type(8)))  float    v8f;

#define BATCH   8
#define SEQ     1024
#define IND     512      // QDIM = KDIM = VDIM
#define NHEADS  8
#define HODIM   64
#define NSLOPE  0.2f
#define PLANE   (BATCH*NHEADS*SEQ)   // one scalar per (b,h,s)

// ---------------------------------------------------------------------------
// Kernel A: a = X[b,s,:]·W[h,:]; store exp(a) and exp(0.2a) as f16 planes.
// (raw 'a' is no longer needed: exp(leaky_relu(s)) == max(e^s, e^{0.2s}))
// ---------------------------------------------------------------------------
__global__ void gat_proj_kernel(const float* __restrict__ X,
                                const float* __restrict__ W,
                                _Float16* __restrict__ e1,
                                _Float16* __restrict__ e2)
{
    int idx = blockIdx.x * blockDim.x + threadIdx.x;
    if (idx >= BATCH * SEQ * NHEADS) return;
    int h = idx & (NHEADS - 1);
    int s = (idx >> 3) & (SEQ - 1);
    int b = idx >> 13;
    const float4* x = (const float4*)(X + ((size_t)b * SEQ + s) * IND);
    const float4* w = (const float4*)(W + (size_t)h * IND);
    float acc = 0.f;
#pragma unroll 4
    for (int i = 0; i < IND / 4; ++i) {
        float4 xv = x[i], wv = w[i];
        acc = fmaf(xv.x, wv.x, acc);
        acc = fmaf(xv.y, wv.y, acc);
        acc = fmaf(xv.z, wv.z, acc);
        acc = fmaf(xv.w, wv.w, acc);
    }
    size_t o = ((size_t)b * NHEADS + h) * SEQ + s;
    e1[o] = (_Float16)__expf(acc);
    e2[o] = (_Float16)__expf(NSLOPE * acc);
}

// ---------------------------------------------------------------------------
// Kernel B: Vproj = V @ WV^T via v_wmma_f32_16x16x32_f16, one wave per
// 16x16 output tile.  Results are scatter-stored in f16 *already arranged as
// the 32x16 B-matrix WMMA fragment layout* kernel C consumes:
//   flat = ((((b*H+h)*(SEQ/32) + kchunk)*4 + ntile)*32 + lane)*16 + elem
// where lane = nc + ((kr/8)%2)*16, elem = kr%8 + (kr/16)*8  (kr = k%32, nc = d%16).
// ---------------------------------------------------------------------------
__global__ void __launch_bounds__(32)
gat_vproj_kernel(const float* __restrict__ V,
                 const float* __restrict__ WV,
                 _Float16* __restrict__ Vp)
{
    int lane = threadIdx.x;
    int rt = blockIdx.x;              // B*SEQ/16 = 512 row tiles (k dimension)
    int ot = blockIdx.y;              // H*64/16  = 32 col tiles (h*64+d)
    int r  = lane & 15;
    int hi = lane >> 4;               // 0 or 1
    int n  = rt * 16 + r;             // global V row   (A-fragment row)
    int o  = ot * 16 + r;             // output feature (B-fragment col)
    const float* vrow = V  + (size_t)n * IND;
    const float* wrow = WV + (size_t)o * IND;     // WV[o, i], i contiguous

    v8f acc = {};
    for (int kc = 0; kc < IND / 32; ++kc) {
        // A fragment (16x32 f16): elems 0..7 -> K = hi*8+0..7,
        //                          elems 8..15 -> K = 16+hi*8+0..7
        float af[16];
        *(float4*)(af + 0)  = *(const float4*)(vrow + kc * 32 + hi * 8);
        *(float4*)(af + 4)  = *(const float4*)(vrow + kc * 32 + hi * 8 + 4);
        *(float4*)(af + 8)  = *(const float4*)(vrow + kc * 32 + 16 + hi * 8);
        *(float4*)(af + 12) = *(const float4*)(vrow + kc * 32 + 16 + hi * 8 + 4);
        // B fragment (32x16 f16): col = lane&15, elems 0..7 -> K = hi*8+0..7, 8..15 -> K = 16+hi*8+0..7
        float bfl[16];
        int i0 = kc * 32 + hi * 8;
        *(float4*)(bfl + 0)  = *(const float4*)(wrow + i0);
        *(float4*)(bfl + 4)  = *(const float4*)(wrow + i0 + 4);
        *(float4*)(bfl + 8)  = *(const float4*)(wrow + i0 + 16);
        *(float4*)(bfl + 12) = *(const float4*)(wrow + i0 + 20);
        v16h afr, bfr;
#pragma unroll
        for (int e = 0; e < 16; ++e) {
            afr[e] = (_Float16)af[e];
            bfr[e] = (_Float16)bfl[e];
        }
        acc = __builtin_amdgcn_wmma_f32_16x16x32_f16(
                  false, afr, false, bfr, (short)0, acc, false, false);
    }
    // C layout: element v at lane L is (M = v + hi*8, N = lane&15)
#pragma unroll
    for (int v = 0; v < 8; ++v) {
        int Ml   = v + hi * 8;
        int nn   = rt * 16 + Ml;
        int bb   = nn >> 10, kseq = nn & 1023;
        int hh   = o >> 6,   d    = o & 63;
        int kc2  = kseq >> 5, kr  = kseq & 31;
        int nt   = d >> 4,    nc  = d & 15;
        int lane2 = nc + (((kr >> 3) & 1) << 4);
        int ee    = (kr & 7) + ((kr >> 4) << 3);
        size_t flat = ((((size_t)(bb * NHEADS + hh) * (SEQ / 32) + kc2) * 4 + nt)
                       * 32 + lane2) * 16 + ee;
        Vp[flat] = (_Float16)acc[v];
    }
}

// ---------------------------------------------------------------------------
// Kernel C: attention.  One wave per (b, h, 16-query tile).
// Score fragment in packed f16:  w = max(eQ*eK, eQ2*eK2)
//   (== exp(leaky_relu(aQ+aK)); exp is monotone, leaky_relu = max(s, 0.2s))
// Softmax denominator via a 5th WMMA against an all-ones B fragment: the
// f32 row-sum lands in every column of accz, already in acc[] row layout.
// ---------------------------------------------------------------------------
__global__ void __launch_bounds__(32)
gat_attn_kernel(const _Float16* __restrict__ eQ1, const _Float16* __restrict__ eQ2,
                const _Float16* __restrict__ eK1, const _Float16* __restrict__ eK2,
                const _Float16* __restrict__ Vp,
                const float* __restrict__ bias,
                float* __restrict__ out)
{
    __shared__ _Float16 sE1[SEQ], sE2[SEQ];
    int lane = threadIdx.x;
    int qt = blockIdx.x;              // SEQ/16 = 64
    int h  = blockIdx.y;
    int b  = blockIdx.z;

    size_t basek = ((size_t)b * NHEADS + h) * SEQ;
    // stage exp(aK) planes into LDS (2 KB each), 16B per lane per pass
    for (int i = lane; i < SEQ / 8; i += 32) {
        ((float4*)sE1)[i] = ((const float4*)(eK1 + basek))[i];
        ((float4*)sE2)[i] = ((const float4*)(eK2 + basek))[i];
    }
    __syncthreads();

    int r  = lane & 15;
    int hi = lane >> 4;
    size_t baseq = basek + qt * 16 + r;
    _Float16 q1 = eQ1[baseq];
    _Float16 q2 = eQ2[baseq];
    v16h q1v, q2v, ones;
#pragma unroll
    for (int e = 0; e < 16; ++e) {
        q1v[e]  = q1;
        q2v[e]  = q2;
        ones[e] = (_Float16)1.0f;
    }

    const _Float16* vpb = Vp + ((size_t)(b * NHEADS + h) * (SEQ / 32)) * 4 * 32 * 16;

    v8f acc[4] = {};
    v8f accz   = {};
    for (int kc = 0; kc < SEQ / 32; ++kc) {
        int r0 = kc * 32 + hi * 8;
        // lane's 16 eK halves: K = r0..r0+7 and r0+16..r0+23 (two b128 reads)
        v16h k1, k2;
        ((float4*)&k1)[0] = *(const float4*)(sE1 + r0);
        ((float4*)&k1)[1] = *(const float4*)(sE1 + r0 + 16);
        ((float4*)&k2)[0] = *(const float4*)(sE2 + r0);
        ((float4*)&k2)[1] = *(const float4*)(sE2 + r0 + 16);

        v16h wfrag = __builtin_elementwise_max(q1v * k1, q2v * k2);

        accz = __builtin_amdgcn_wmma_f32_16x16x32_f16(
                   false, wfrag, false, ones, (short)0, accz, false, false);
#pragma unroll
        for (int nt = 0; nt < 4; ++nt) {
            const v16h bfr = *(const v16h*)(vpb + ((((size_t)kc * 4 + nt) * 32 + lane) * 16));
            acc[nt] = __builtin_amdgcn_wmma_f32_16x16x32_f16(
                          false, wfrag, false, bfr, (short)0, acc[nt], false, false);
        }
    }

    float rinv[8];
#pragma unroll
    for (int v = 0; v < 8; ++v)
        rinv[v] = 1.0f / accz[v];     // row (v + hi*8) sum, same layout as acc

#pragma unroll
    for (int nt = 0; nt < 4; ++nt) {
        int d = nt * 16 + r;
        float bi = bias[h * HODIM + d];
#pragma unroll
        for (int v = 0; v < 8; ++v) {
            int qq = qt * 16 + v + hi * 8;
            out[((size_t)b * SEQ + qq) * (NHEADS * HODIM) + h * HODIM + d] =
                acc[nt][v] * rinv[v] + bi;
        }
    }
}

// ---------------------------------------------------------------------------
extern "C" void kernel_launch(void* const* d_in, const int* in_sizes, int n_in,
                              void* d_out, int out_size, void* d_ws, size_t ws_size,
                              hipStream_t stream)
{
    (void)in_sizes; (void)n_in; (void)out_size; (void)ws_size;
    const float* Q    = (const float*)d_in[0];
    const float* K    = (const float*)d_in[1];
    const float* V    = (const float*)d_in[2];
    const float* WQ   = (const float*)d_in[3];
    const float* WK   = (const float*)d_in[4];
    const float* WV   = (const float*)d_in[5];
    const float* bias = (const float*)d_in[6];
    float* out = (float*)d_out;

    // workspace: 4 f16 exp planes (128 KB each) + packed f16 Vproj (8 MB)
    _Float16* hw  = (_Float16*)d_ws;
    _Float16* eQ1 = hw + 0 * PLANE;
    _Float16* eQ2 = hw + 1 * PLANE;
    _Float16* eK1 = hw + 2 * PLANE;
    _Float16* eK2 = hw + 3 * PLANE;
    _Float16* Vp  = hw + 4 * PLANE;

    int nproj = BATCH * SEQ * NHEADS;
    gat_proj_kernel<<<nproj / 256, 256, 0, stream>>>(Q, WQ, eQ1, eQ2);
    gat_proj_kernel<<<nproj / 256, 256, 0, stream>>>(K, WK, eK1, eK2);
    gat_vproj_kernel<<<dim3(BATCH * SEQ / 16, NHEADS * HODIM / 16), 32, 0, stream>>>(V, WV, Vp);
    gat_attn_kernel<<<dim3(SEQ / 16, NHEADS, BATCH), 32, 0, stream>>>(
        eQ1, eQ2, eK1, eK2, Vp, bias, out);
}
